// PDG2Seq_74878459838635
// MI455X (gfx1250) — matrix operand order repaired and checked by reading, not compiled, not run on hardware
//
#include <hip/hip_runtime.h>
#include <math.h>

typedef __attribute__((ext_vector_type(16))) __bf16 v16bf;
typedef __attribute__((ext_vector_type(8)))  __bf16 v8bf;
typedef __attribute__((ext_vector_type(8)))  float  v8f;

// ---- problem constants ----
constexpr int BB  = 32;            // batch
constexpr int TT  = 12;            // encoder steps
constexpr int HH  = 12;            // decoder steps (HOR)
constexpr int NN  = 883;           // nodes
constexpr int DD  = 10;            // embed dim
constexpr int HD  = 64;            // rnn units
constexpr int CIc = 65;            // 1 + HID == OUT + HID
constexpr int CIP = 68;            // padded xs/agg row stride (16B-aligned rows)
constexpr int COG = 128;           // gate output (2*HID)
constexpr int COU = 64;            // update output (HID)
constexpr int KKc = DD * 2 * CIc;  // 1300 folded K dim
constexpr int KKB = KKc + DD;      // 1310: +10 bias rows (z = nv1 there)
constexpr int KKP = 1312;          // padded to 41*32
constexpr int NKC = KKP / 32;      // 41 K-chunks for adaptive GEMM
constexpr int NTc = (NN + 15) / 16;// 56 row tiles
constexpr int NPc = NTc * 16;      // 896 padded node count
constexpr int MCc = NPc / 32;      // 28 key chunks for attention PV
constexpr int BNc = BB * NN;       // 28256

static __device__ __forceinline__ v16bf join8(v8bf a, v8bf b) {
  return __builtin_shufflevector(a, b, 0,1,2,3,4,5,6,7,8,9,10,11,12,13,14,15);
}

// ---------------- elementwise kernels ----------------
__global__ void k_zero(float* __restrict__ h, float* __restrict__ go) {
  int i = blockIdx.x * blockDim.x + threadIdx.x;
  if (i < BNc * HD) h[i] = 0.f;
  if (i < BNc)      go[i] = 0.f;
}

// Fold Wp[d,k,i,o] + bias bp[d,o] (f32) into bf16 weights laid out in exact
// A-fragment order: wf[(((kc*2 + run)*2 + hi)*CO + o)*8 + e], kk = kc*32 + run*16 + hi*8 + e.
// Rows kk in [1300,1310) hold the bias; [1310,1312) are zero.
__global__ void k_fold(const float* __restrict__ w, const float* __restrict__ bp,
                       __bf16* __restrict__ wf, int CO) {
  int i = blockIdx.x * blockDim.x + threadIdx.x;
  if (i >= KKP * CO) return;
  int e  = i & 7;
  int t  = i >> 3;
  int o  = t % CO;
  int u  = t / CO;
  int hi = u & 1;
  int v2 = u >> 1;
  int run = v2 & 1;
  int kc  = v2 >> 1;
  int kk  = kc * 32 + run * 16 + hi * 8 + e;
  float val = 0.f;
  if (kk < KKc)      val = w[(size_t)kk * CO + o];
  else if (kk < KKB) val = bp[(kk - KKc) * CO + o];
  wf[i] = (__bf16)val;
}

// nv1/nv2 = tanh(E * tid_emb * diw_emb),  xs = concat(x, h) with zeroed pads
__global__ void k_embed(const int* __restrict__ tid, const int* __restrict__ diw,
                        const float* __restrict__ te1, const float* __restrict__ te2,
                        const float* __restrict__ de1, const float* __restrict__ de2,
                        const float* __restrict__ nodeE,
                        const float* __restrict__ xsrc, int enc, int t, int Tdim,
                        const float* __restrict__ h,
                        float* __restrict__ nv1, float* __restrict__ nv2,
                        float* __restrict__ xs) {
  int bn = blockIdx.x * blockDim.x + threadIdx.x;
  if (bn >= BNc) return;
  int b = bn / NN, n = bn - b * NN;
  int ti = tid[((size_t)b * Tdim + t) * NN + n];
  int wi = diw[((size_t)b * Tdim + t) * NN + n];
#pragma unroll
  for (int d = 0; d < DD; ++d) {
    float g = nodeE[n * DD + d];
    nv1[(size_t)bn * DD + d] = tanhf(g * te1[ti * DD + d] * de1[wi * DD + d]);
    nv2[(size_t)bn * DD + d] = tanhf(g * te2[ti * DD + d] * de2[wi * DD + d]);
  }
  float x = enc ? xsrc[((size_t)b * Tdim + t) * NN + n] : xsrc[bn];
  float* xr = xs + (size_t)bn * CIP;
  xr[0] = x;
#pragma unroll 8
  for (int j = 0; j < HD; ++j) xr[1 + j] = h[(size_t)bn * HD + j];
  xr[65] = 0.f; xr[66] = 0.f; xr[67] = 0.f;
}

// xs2 = concat(x, z*h)   (z = zr[..., :64])
__global__ void k_cand(const float* __restrict__ xsrc, int enc, int t, int Tdim,
                       const float* __restrict__ zr, const float* __restrict__ h,
                       float* __restrict__ xs) {
  int bn = blockIdx.x * blockDim.x + threadIdx.x;
  if (bn >= BNc) return;
  int b = bn / NN, n = bn - b * NN;
  float x = enc ? xsrc[((size_t)b * Tdim + t) * NN + n] : xsrc[bn];
  float* xr = xs + (size_t)bn * CIP;
  xr[0] = x;
#pragma unroll 8
  for (int j = 0; j < HD; ++j)
    xr[1 + j] = zr[(size_t)bn * COG + j] * h[(size_t)bn * HD + j];
  xr[65] = 0.f; xr[66] = 0.f; xr[67] = 0.f;
}

// h = r*h + (1-r)*hc ; decoder also go = h @ projW^T + b, write output slice
__global__ void k_updh(const float* __restrict__ zr, const float* __restrict__ hc,
                       float* __restrict__ h, float* __restrict__ go,
                       const float* __restrict__ pw, const float* __restrict__ pb,
                       float* __restrict__ out, int dec, int t) {
  int bn = blockIdx.x * blockDim.x + threadIdx.x;
  if (bn >= BNc) return;
  float s = 0.f;
#pragma unroll 8
  for (int j = 0; j < HD; ++j) {
    float r  = zr[(size_t)bn * COG + HD + j];
    float hn = r * h[(size_t)bn * HD + j] + (1.f - r) * hc[(size_t)bn * HD + j];
    h[(size_t)bn * HD + j] = hn;
    s += hn * pw[j];
  }
  if (dec) {
    float g = s + pb[0];
    go[bn] = g;
    int b = bn / NN, n = bn - b * NN;
    out[((size_t)b * HH + t) * NN + n] = g;
  }
}

// ---------------- fused adjacency + aggregation ----------------
// One wave per (b, 16-query tile). Pass1: logits via WMMA (Q 16xK, K padded 10->32),
// exp(relu) kept in LDS (bf16) + row sums. Pass2: Agg^T = V^T x P^T via WMMA,
// vectorized float4 output stores.
__global__ void __launch_bounds__(32)
k_attn(const float* __restrict__ nv1, const float* __restrict__ nv2,
       const float* __restrict__ xv, float* __restrict__ agg) {
  __shared__ __align__(16) __bf16 sP[16][NPc];   // 28KB
  __shared__ float sQ[16][12];
  __shared__ float sRS[16];
  __shared__ float sInv[16];
  const int L = threadIdx.x;
  const int b = blockIdx.y, nt = blockIdx.x;
  const int col = L & 15, hi = L >> 4;

  for (int idx = L; idx < 16 * DD; idx += 32) {
    int r = idx / DD, d = idx - r * DD;
    int n = nt * 16 + r;
    int ok = (n < NN);
    sQ[r][d] = nv1[((size_t)b * NN + (ok ? n : 0)) * DD + d] * (ok ? 1.f : 0.f);
  }
  if (L < 16) sRS[L] = 0.f;
  __syncthreads();

  // Q fragment (A operand): row = query col index, K = d (zero-padded past DD)
  v16bf qa;
#pragma unroll
  for (int e = 0; e < 16; ++e) {
    int K = hi * 16 + e;
    float qv = sQ[col][(K < DD) ? K : 0];
    qa[e] = (__bf16)((K < DD) ? qv : 0.f);
  }

  const v8f zC = (v8f){0.f,0.f,0.f,0.f,0.f,0.f,0.f,0.f};
  float psum[8];
#pragma unroll
  for (int j = 0; j < 8; ++j) psum[j] = 0.f;

  // pass 1: 56 key tiles of 16
  for (int mt = 0; mt < NTc; ++mt) {
    int m   = mt * 16 + col;
    int mok = (m < NN);
    const float* kp = nv2 + ((size_t)b * NN + (mok ? m : 0)) * DD;
    float msk = (hi == 0 && mok) ? 1.f : 0.f;
    float kv[DD];
#pragma unroll
    for (int d = 0; d < DD; ++d) kv[d] = msk * kp[d];   // unconditional loads
    v16bf kb;
#pragma unroll
    for (int e = 0; e < 16; ++e) kb[e] = (e < DD) ? (__bf16)kv[e] : (__bf16)0.f;
    v8f lg = __builtin_amdgcn_wmma_f32_16x16x32_bf16(false, qa, false, kb,
                                                     (short)0, zC, false, false);
    float cm = mok ? 1.f : 0.f;
#pragma unroll
    for (int j = 0; j < 8; ++j) {
      float ev = cm * __expf(fmaxf(lg[j], 0.f));   // exp(relu(s)); 0 for padded keys
      psum[j] += ev;
      sP[j + hi * 8][m] = (__bf16)ev;
    }
  }
#pragma unroll
  for (int j = 0; j < 8; ++j) atomicAdd(&sRS[j + hi * 8], psum[j]);  // ds_add_f32
  __syncthreads();
  if (L < 16) sInv[L] = 1.f / sRS[L];
  __syncthreads();

  // pass 2: Agg^T = V^T (A) x P^T (B); D rows = feature c, cols = query n
  v8f acc[5];
#pragma unroll
  for (int ct = 0; ct < 5; ++ct) acc[ct] = zC;

  for (int mc = 0; mc < MCc; ++mc) {
    const __bf16* prow = &sP[col][mc * 32 + hi * 16];
    v16bf pb = join8(*(const v8bf*)prow, *(const v8bf*)(prow + 8));  // 2x ds_load_b128
#pragma unroll
    for (int ct = 0; ct < 5; ++ct) {
      int c   = ct * 16 + col;
      int cok = (c < CIc);
      int cc  = cok ? c : 0;
      v16bf av;
#pragma unroll
      for (int e = 0; e < 16; ++e) {
        int kl  = hi * 16 + e;
        int m   = mc * 32 + kl;
        int mok = (m < NN);
        float v = xv[((size_t)b * NN + (mok ? m : 0)) * CIP + cc];   // coalesced b32
        av[e] = (__bf16)((cok && mok) ? v : 0.f);
      }
      acc[ct] = __builtin_amdgcn_wmma_f32_16x16x32_bf16(false, av, false, pb,
                                                        (short)0, acc[ct], false, false);
    }
  }

  // epilogue: per-lane query n fixed, feature c contiguous -> float4 stores
  int n = nt * 16 + col;
  if (n < NN) {
    float inv = sInv[col];
    float* orow = agg + ((size_t)b * NN + n) * CIP;
#pragma unroll
    for (int ct = 0; ct < 4; ++ct) {
      int cb = ct * 16 + hi * 8;
      *(float4*)(orow + cb)     = make_float4(acc[ct][0]*inv, acc[ct][1]*inv,
                                              acc[ct][2]*inv, acc[ct][3]*inv);
      *(float4*)(orow + cb + 4) = make_float4(acc[ct][4]*inv, acc[ct][5]*inv,
                                              acc[ct][6]*inv, acc[ct][7]*inv);
    }
    if (hi == 0)  // c=64 valid, zero the pads 65..67
      *(float4*)(orow + 64) = make_float4(acc[4][0]*inv, 0.f, 0.f, 0.f);
  }
}

// ---------------- adaptive-weight GEMM ----------------
// out^T = Wfold (A, fragment-order global) x z^T (B, generated on the fly).
// z[kk<1300] = nv1[d]*xg[k,i]; z[1300+d] = nv1[d] (bias rows). Vectorized stores.
template <int OT, int CO, int ACT>
__global__ void __launch_bounds__(32)
k_gemm(const float* __restrict__ xs, const float* __restrict__ agg,
       const float* __restrict__ nv1, const __bf16* __restrict__ wf,
       float* __restrict__ out) {
  __shared__ __align__(16) float sX[16][CIP];
  __shared__ __align__(16) float sG[16][CIP];
  __shared__ float sQ[16][12];
  const int L = threadIdx.x;
  const int b = blockIdx.y, nt = blockIdx.x;
  const int col = L & 15, hi = L >> 4;

  for (int idx = L; idx < 16 * (CIP / 4); idx += 32) {   // float4 staging
    int r = idx / (CIP / 4), q = idx - r * (CIP / 4);
    int n = nt * 16 + r;
    int ok = (n < NN);
    size_t rb = ((size_t)b * NN + (ok ? n : 0)) * CIP;
    float mk = ok ? 1.f : 0.f;
    float4 vx = ((const float4*)(xs + rb))[q];
    float4 vg = ((const float4*)(agg + rb))[q];
    ((float4*)&sX[r][0])[q] = make_float4(vx.x*mk, vx.y*mk, vx.z*mk, vx.w*mk);
    ((float4*)&sG[r][0])[q] = make_float4(vg.x*mk, vg.y*mk, vg.z*mk, vg.w*mk);
  }
  for (int idx = L; idx < 16 * DD; idx += 32) {
    int r = idx / DD, d = idx - r * DD;
    int n = nt * 16 + r;
    int ok = (n < NN);
    sQ[r][d] = nv1[((size_t)b * NN + (ok ? n : 0)) * DD + d] * (ok ? 1.f : 0.f);
  }
  __syncthreads();

  v8f acc[OT];
#pragma unroll
  for (int ot = 0; ot < OT; ++ot) acc[ot] = (v8f){0.f,0.f,0.f,0.f,0.f,0.f,0.f,0.f};

  for (int kc = 0; kc < NKC; ++kc) {
    if (kc + 1 < NKC)
      __builtin_prefetch(wf + (size_t)(kc + 1) * 4 * CO * 8, 0, 1);  // global_prefetch
    // B fragment: z^T column = this lane's node row, K = kk (generated)
    v16bf zb;
#pragma unroll
    for (int e = 0; e < 16; ++e) {
      int kk = kc * 32 + ((e < 8) ? (hi * 8 + e) : (16 + hi * 8 + (e - 8)));
      int kz = (kk < KKc) ? kk : 0;
      int d  = kz / 130;
      int r2 = kz - d * 130;
      int k  = r2 / CIc;
      int i  = r2 - k * CIc;
      float zv = sQ[col][d] * (k ? sG[col][i] : sX[col][i]);
      int bi = kk - KKc;
      float bv = sQ[col][(bi >= 0 && bi < DD) ? bi : 0];
      float v  = (kk < KKc) ? zv : ((kk < KKB) ? bv : 0.f);
      zb[e] = (__bf16)v;
    }
#pragma unroll
    for (int ot = 0; ot < OT; ++ot) {
      int o = ot * 16 + col;
      const v8bf* p0 = (const v8bf*)(wf + ((size_t)(((kc * 2 + 0) * 2 + hi) * CO + o) << 3));
      const v8bf* p1 = (const v8bf*)(wf + ((size_t)(((kc * 2 + 1) * 2 + hi) * CO + o) << 3));
      v16bf wa = join8(*p0, *p1);   // 2x coalesced global_load_b128
      acc[ot] = __builtin_amdgcn_wmma_f32_16x16x32_bf16(false, wa, false, zb,
                                                        (short)0, acc[ot], false, false);
    }
  }

  // epilogue: per-lane node n fixed, outputs o contiguous -> float4 stores
  int n = nt * 16 + col;
  if (n < NN) {
    float* orow = out + ((size_t)b * NN + n) * CO;
#pragma unroll
    for (int ot = 0; ot < OT; ++ot) {
      int ob = ot * 16 + hi * 8;
      float v[8];
#pragma unroll
      for (int j = 0; j < 8; ++j) {
        float x = acc[ot][j];
        v[j] = ACT ? tanhf(x) : (1.f / (1.f + __expf(-x)));
      }
      *(float4*)(orow + ob)     = make_float4(v[0], v[1], v[2], v[3]);
      *(float4*)(orow + ob + 4) = make_float4(v[4], v[5], v[6], v[7]);
    }
  }
}

// ---------------- launcher ----------------
extern "C" void kernel_launch(void* const* d_in, const int* in_sizes, int n_in,
                              void* d_out, int out_size, void* d_ws, size_t ws_size,
                              hipStream_t stream) {
  const float* src_vals = (const float*)d_in[0];
  const float* node_emb = (const float*)d_in[1];
  const float* te1 = (const float*)d_in[2];
  const float* te2 = (const float*)d_in[3];
  const float* de1 = (const float*)d_in[4];
  const float* de2 = (const float*)d_in[5];
  const float* egW = (const float*)d_in[6];
  const float* egb = (const float*)d_in[7];
  const float* euW = (const float*)d_in[8];
  const float* eub = (const float*)d_in[9];
  const float* dgW = (const float*)d_in[10];
  const float* dgb = (const float*)d_in[11];
  const float* duW = (const float*)d_in[12];
  const float* dub = (const float*)d_in[13];
  const float* pw  = (const float*)d_in[14];
  const float* pb  = (const float*)d_in[15];
  const int* s_tid = (const int*)d_in[16];
  const int* s_diw = (const int*)d_in[17];
  const int* t_tid = (const int*)d_in[18];
  const int* t_diw = (const int*)d_in[19];
  float* out = (float*)d_out;

  char* p = (char*)d_ws;
  auto alloc = [&](size_t bytes) { char* r = p; p += (bytes + 255) & ~(size_t)255; return r; };
  float*  h   = (float*)alloc((size_t)BNc * HD * 4);
  float*  go  = (float*)alloc((size_t)BNc * 4);
  float*  nv1 = (float*)alloc((size_t)BNc * DD * 4);
  float*  nv2 = (float*)alloc((size_t)BNc * DD * 4);
  float*  xs  = (float*)alloc((size_t)BNc * CIP * 4);
  float*  agg = (float*)alloc((size_t)BNc * CIP * 4);
  float*  zr  = (float*)alloc((size_t)BNc * COG * 4);
  float*  hc  = (float*)alloc((size_t)BNc * HD * 4);
  __bf16* wge = (__bf16*)alloc((size_t)COG * KKP * 2);
  __bf16* wue = (__bf16*)alloc((size_t)COU * KKP * 2);
  __bf16* wgd = (__bf16*)alloc((size_t)COG * KKP * 2);
  __bf16* wud = (__bf16*)alloc((size_t)COU * KKP * 2);

  dim3 wg(NTc, BB), wave(32);
  int ebl = (BNc + 255) / 256;

  k_zero<<<(BNc * HD + 255) / 256, 256, 0, stream>>>(h, go);
  k_fold<<<(COG * KKP + 255) / 256, 256, 0, stream>>>(egW, egb, wge, COG);
  k_fold<<<(COU * KKP + 255) / 256, 256, 0, stream>>>(euW, eub, wue, COU);
  k_fold<<<(COG * KKP + 255) / 256, 256, 0, stream>>>(dgW, dgb, wgd, COG);
  k_fold<<<(COU * KKP + 255) / 256, 256, 0, stream>>>(duW, dub, wud, COU);

  for (int t = 0; t < TT; ++t) {   // encoder
    k_embed<<<ebl, 256, 0, stream>>>(s_tid, s_diw, te1, te2, de1, de2, node_emb,
                                     src_vals, 1, t, TT, h, nv1, nv2, xs);
    k_attn<<<wg, wave, 0, stream>>>(nv1, nv2, xs, agg);
    k_gemm<8, COG, 0><<<wg, wave, 0, stream>>>(xs, agg, nv1, wge, zr);
    k_cand<<<ebl, 256, 0, stream>>>(src_vals, 1, t, TT, zr, h, xs);
    k_attn<<<wg, wave, 0, stream>>>(nv1, nv2, xs, agg);
    k_gemm<4, COU, 1><<<wg, wave, 0, stream>>>(xs, agg, nv1, wue, hc);
    k_updh<<<ebl, 256, 0, stream>>>(zr, hc, h, go, pw, pb, out, 0, t);
  }
  for (int t = 0; t < HH; ++t) {   // decoder
    k_embed<<<ebl, 256, 0, stream>>>(t_tid, t_diw, te1, te2, de1, de2, node_emb,
                                     go, 0, t, HH, h, nv1, nv2, xs);
    k_attn<<<wg, wave, 0, stream>>>(nv1, nv2, xs, agg);
    k_gemm<8, COG, 0><<<wg, wave, 0, stream>>>(xs, agg, nv1, wgd, zr);
    k_cand<<<ebl, 256, 0, stream>>>(go, 0, t, HH, zr, h, xs);
    k_attn<<<wg, wave, 0, stream>>>(nv1, nv2, xs, agg);
    k_gemm<4, COU, 1><<<wg, wave, 0, stream>>>(xs, agg, nv1, wud, hc);
    k_updh<<<ebl, 256, 0, stream>>>(zr, hc, h, go, pw, pb, out, 1, t);
  }
  (void)in_sizes; (void)n_in; (void)out_size; (void)ws_size;
}
